// Seq2Seq_3143916061083
// MI455X (gfx1250) — matrix operand order repaired and checked by reading, not compiled, not run on hardware
//
#include <hip/hip_runtime.h>
#include <cmath>

#define B_   16
#define LS_  128
#define LT_  64
#define E_   512
#define H_   512
#define G4_  2048      // 4*H
#define VT_  32000
#define PAD_ 3

typedef __attribute__((ext_vector_type(16))) __bf16 v16bf;
typedef __attribute__((ext_vector_type(8)))  float  v8f;
typedef unsigned short ush;

// ---------------- WMMA helpers (layouts per CDNA5 ISA 7.12.2) ----------------

__device__ __forceinline__ __bf16 f2b(float f) { return (__bf16)f; }

// A-matrix 16x32 bf16: lane m=lane&15, g=lane>>4. half i -> K:
// i<8 : K = 8*g + i ; i>=8: K = 16 + 8*g + (i-8)
__device__ __forceinline__ int a_kidx(int i, int g) {
  return (i < 8) ? (8 * g + i) : (8 + 8 * g + i);
}

__device__ __forceinline__ v16bf loadA_f32(const float* base, int ld, int k0, int lane) {
  int m = lane & 15, g = lane >> 4;
  v16bf a;
#pragma unroll
  for (int i = 0; i < 16; ++i) a[i] = f2b(base[(size_t)m * ld + k0 + a_kidx(i, g)]);
  return a;
}

__device__ __forceinline__ v16bf loadA_tok(const float* emb, const int* seq, int seqlen,
                                           int t, int k0, int lane) {
  int m = lane & 15, g = lane >> 4;
  const float* row = emb + (size_t)seq[m * seqlen + t] * E_;
  v16bf a;
#pragma unroll
  for (int i = 0; i < 16; ++i) a[i] = f2b(row[k0 + a_kidx(i, g)]);
  return a;
}

// Packed tile layout (B and pre-packed A): 512 bf16 per tile (1024B), lane's 16
// halves contiguous at lane*16 -> one aligned v16bf load = 2 x global_load_b128.
__device__ __forceinline__ const __bf16* tileB_ptr(const ush* P, int nKt, int kt,
                                                   int nt, int lane) {
  return (const __bf16*)P + (((size_t)nt * nKt + kt) << 9) + lane * 16;
}
__device__ __forceinline__ v16bf loadB_pk(const ush* P, int nKt, int kt, int nt,
                                          int lane) {
  return *(const v16bf*)tileB_ptr(P, nKt, kt, nt, lane);
}
__device__ __forceinline__ v16bf loadA_pk(const ush* P, int kt, int lane) {
  return *(const v16bf*)((const __bf16*)P + ((size_t)kt << 9) + lane * 16);
}

__device__ __forceinline__ v8f wmma_bf(v16bf a, v16bf b, v8f c) {
  return __builtin_amdgcn_wmma_f32_16x16x32_bf16(false, a, false, b, (short)0, c,
                                                 false, false);
}

// C/D 16x16 f32: lane n=lane&15, VGPR r -> M = r + 8*(lane>>4)
__device__ __forceinline__ void storeD(float* base, int ldn, v8f c, int lane) {
  int n = lane & 15, mb = (lane >> 4) * 8;
#pragma unroll
  for (int r = 0; r < 8; ++r) base[(size_t)(mb + r) * ldn + n] = c[r];
}

__device__ __forceinline__ float sigm(float x) { return 1.f / (1.f + __expf(-x)); }

// Branch-free pipelined GEMM segments (single A source each, acc carried across)
__device__ __forceinline__ v8f seg_f32(v8f acc, const float* A, int ld, const ush* Bp,
                                       int nKtB, int ktB0, int nk, int nt, int lane) {
  v16bf a0 = loadA_f32(A, ld, 0, lane);
  v16bf b0 = loadB_pk(Bp, nKtB, ktB0, nt, lane);
  for (int kt = 0; kt < nk - 1; ++kt) {
    v16bf a1 = loadA_f32(A, ld, (kt + 1) * 32, lane);
    v16bf b1 = loadB_pk(Bp, nKtB, ktB0 + kt + 1, nt, lane);
    acc = wmma_bf(a0, b0, acc);
    a0 = a1; b0 = b1;
  }
  return wmma_bf(a0, b0, acc);
}

__device__ __forceinline__ v8f seg_tok(v8f acc, const float* emb, const int* seq,
                                       int seqlen, int t, const ush* Bp, int nKtB,
                                       int ktB0, int nk, int nt, int lane) {
  v16bf a0 = loadA_tok(emb, seq, seqlen, t, 0, lane);
  v16bf b0 = loadB_pk(Bp, nKtB, ktB0, nt, lane);
  for (int kt = 0; kt < nk - 1; ++kt) {
    v16bf a1 = loadA_tok(emb, seq, seqlen, t, (kt + 1) * 32, lane);
    v16bf b1 = loadB_pk(Bp, nKtB, ktB0 + kt + 1, nt, lane);
    acc = wmma_bf(a0, b0, acc);
    a0 = a1; b0 = b1;
  }
  return wmma_bf(a0, b0, acc);
}

// ---------------- utility kernels ----------------

__global__ void k_zero(float* p, int n) {
  int i = blockIdx.x * blockDim.x + threadIdx.x;
  if (i < n) p[i] = 0.f;
}

// Convert W[K][N] f32 row-major -> packed bf16 WMMA-B tile layout.
__global__ void k_pack(const float* W, ush* P, int K, int N) {
  int idx = blockIdx.x * blockDim.x + threadIdx.x;
  if (idx >= K * N) return;
  int tile = idx >> 9, within = idx & 511;
  int lane = within >> 4, i = within & 15;
  int nKt = K >> 5;
  int nt = tile / nKt, kt = tile - nt * nKt;
  int g = lane >> 4, n = lane & 15;
  int k = kt * 32 + a_kidx(i, g);
  int nn = nt * 16 + n;
  ((__bf16*)P)[idx] = f2b(W[(size_t)k * N + nn]);
}

// Pack [h(16x512), ctx(16x512)] f32 -> 32 bf16 WMMA-A fragment tiles (32KB).
__global__ void k_packA_hx(const float* h, const float* ctx, ush* P) {
  int idx = blockIdx.x * blockDim.x + threadIdx.x;   // 32 tiles * 512
  if (idx >= 32 * 512) return;
  int kt = idx >> 9, within = idx & 511;
  int lane = within >> 4, i = within & 15;
  int g = lane >> 4, m = lane & 15;
  int k = kt * 32 + a_kidx(i, g);
  float v = (k < H_) ? h[m * H_ + k] : ctx[m * H_ + (k - H_)];
  ((__bf16*)P)[idx] = f2b(v);
}

// ---------------- encoder LSTM step ----------------
// grid: H/16 blocks, 128 threads (4 waves; wave w computes gate w's 16x16 z-tile)
__global__ __launch_bounds__(128)
void k_enc_step(const int* src, const float* emb, const ush* Wx, const ush* Wh,
                const float* bias, const float* h_in, float* h_out, float* c_st,
                float* enc_hs, int t) {
  __shared__ float z[4][16][16];
  int tid = threadIdx.x, lane = tid & 31, w = tid >> 5;
  int n0 = blockIdx.x * 16;
  int nt = w * (H_ / 16) + blockIdx.x;     // column tile within 4H
  v8f acc = {};
  acc = seg_tok(acc, emb, src, LS_, t, Wx, 16, 0, 16, nt, lane);
  acc = seg_f32(acc, h_in, H_, Wh, 16, 0, 16, nt, lane);
  int n = lane & 15, mb = (lane >> 4) * 8;
  float bv = bias[w * H_ + n0 + n];
#pragma unroll
  for (int r = 0; r < 8; ++r) z[w][mb + r][n] = acc[r] + bv;
  __syncthreads();
  for (int e = tid; e < 256; e += 128) {
    int m = e >> 4, j = e & 15, hc = n0 + j;
    float ig = sigm(z[0][m][j]), fg = sigm(z[1][m][j]);
    float gg = tanhf(z[2][m][j]), og = sigm(z[3][m][j]);
    float cn = fg * c_st[m * H_ + hc] + ig * gg;
    float hn = og * tanhf(cn);
    c_st[m * H_ + hc] = cn;
    h_out[m * H_ + hc] = hn;
    enc_hs[(size_t)t * B_ * H_ + m * H_ + hc] = hn;   // [Ls][B][H]
  }
}

// ---------------- decoder LSTM step (x = [embed, context]) ----------------
__global__ __launch_bounds__(128)
void k_dec_step(const int* tgt, const float* emb, const float* ctx, const ush* Wx,
                const ush* Wh, const float* bias, const float* h_in, float* h_out,
                float* c_st, int t) {
  __shared__ float z[4][16][16];
  int tid = threadIdx.x, lane = tid & 31, w = tid >> 5;
  int n0 = blockIdx.x * 16;
  int nt = w * (H_ / 16) + blockIdx.x;
  v8f acc = {};
  acc = seg_tok(acc, emb, tgt, LT_, t, Wx, 32, 0, 16, nt, lane);
  acc = seg_f32(acc, ctx, H_, Wx, 32, 16, 16, nt, lane);
  acc = seg_f32(acc, h_in, H_, Wh, 16, 0, 16, nt, lane);
  int n = lane & 15, mb = (lane >> 4) * 8;
  float bv = bias[w * H_ + n0 + n];
#pragma unroll
  for (int r = 0; r < 8; ++r) z[w][mb + r][n] = acc[r] + bv;
  __syncthreads();
  for (int e = tid; e < 256; e += 128) {
    int m = e >> 4, j = e & 15, hc = n0 + j;
    float ig = sigm(z[0][m][j]), fg = sigm(z[1][m][j]);
    float gg = tanhf(z[2][m][j]), og = sigm(z[3][m][j]);
    float cn = fg * c_st[m * H_ + hc] + ig * gg;
    float hn = og * tanhf(cn);
    c_st[m * H_ + hc] = cn;
    h_out[m * H_ + hc] = hn;
  }
}

// ---------------- generic GEMM: C[M,N] = A[M,K] (f32) x packed-B (bf16) ----------------
__global__ __launch_bounds__(128)
void k_mm(const float* A, const ush* B, float* C, int M, int N, int K) {
  int lane = threadIdx.x & 31, w = threadIdx.x >> 5;
  int mt = blockIdx.x;
  int nt = blockIdx.y * 4 + w;
  if (nt * 16 >= N) return;   // wave-uniform
  const float* Ab = A + (size_t)mt * 16 * K;
  int nKt = K >> 5;
  v8f acc = {};
  acc = seg_f32(acc, Ab, K, B, nKt, 0, nKt, nt, lane);
  storeD(C + (size_t)mt * 16 * N + nt * 16, N, acc, lane);
}

// ---------------- additive attention: one block per batch row ----------------
__global__ __launch_bounds__(128)
void k_attn(const int* src, const float* kp, const float* q, const float* vattn,
            const float* enc_hs, float* ctx) {
  __shared__ float sc[LS_];
  __shared__ float red[LS_];
  int b = blockIdx.x, tid = threadIdx.x, lane = tid & 31, w = tid >> 5;
  for (int l = w; l < LS_; l += 4) {
    const float* kpl = kp + ((size_t)l * B_ + b) * H_;
    const float* qb = q + b * H_;
    float s = 0.f;
    for (int j = lane; j < H_; j += 32) s += vattn[j] * tanhf(kpl[j] + qb[j]);
#pragma unroll
    for (int off = 16; off > 0; off >>= 1) s += __shfl_down(s, off, 32);
    if (lane == 0) sc[l] = (src[b * LS_ + l] != PAD_) ? s : -INFINITY;
  }
  __syncthreads();
  float val = sc[tid];
  red[tid] = val; __syncthreads();
  for (int s2 = 64; s2 > 0; s2 >>= 1) {
    if (tid < s2) red[tid] = fmaxf(red[tid], red[tid + s2]);
    __syncthreads();
  }
  float mx = red[0]; __syncthreads();
  float e = __expf(val - mx);
  red[tid] = e; __syncthreads();
  for (int s2 = 64; s2 > 0; s2 >>= 1) {
    if (tid < s2) red[tid] += red[tid + s2];
    __syncthreads();
  }
  float denom = red[0]; __syncthreads();
  sc[tid] = e / denom;   // alpha
  __syncthreads();
  for (int hc = tid; hc < H_; hc += 128) {
    float a = 0.f;
    for (int l = 0; l < LS_; ++l) a += sc[l] * enc_hs[((size_t)l * B_ + b) * H_ + hc];
    ctx[b * H_ + hc] = a;
  }
}

// ---------------- output projection: packed-A[16,1024] x Wout[1024,32000] ----------------
// 2 N-tiles per wave; one A fragment feeds two wmmas per K-step.
__global__ __launch_bounds__(128)
void k_logits(const ush* Apk, const ush* Wout, const float* bout, float* logits) {
  int lane = threadIdx.x & 31, w = threadIdx.x >> 5;
  int nt0 = (blockIdx.x * 4 + w) * 2;
  const int NK = 32;  // K = 1024
  v8f acc0 = {}, acc1 = {};
  v16bf a  = loadA_pk(Apk, 0, lane);
  v16bf b0 = loadB_pk(Wout, NK, 0, nt0, lane);
  v16bf b1 = loadB_pk(Wout, NK, 0, nt0 + 1, lane);
  for (int kt = 0; kt < NK - 1; ++kt) {
    v16bf an  = loadA_pk(Apk, kt + 1, lane);
    v16bf bn0 = loadB_pk(Wout, NK, kt + 1, nt0, lane);
    v16bf bn1 = loadB_pk(Wout, NK, kt + 1, nt0 + 1, lane);
    acc0 = wmma_bf(a, b0, acc0);
    acc1 = wmma_bf(a, b1, acc1);
    a = an; b0 = bn0; b1 = bn1;
  }
  acc0 = wmma_bf(a, b0, acc0);
  acc1 = wmma_bf(a, b1, acc1);
  int n = lane & 15, mb = (lane >> 4) * 8;
  float bv0 = bout[nt0 * 16 + n], bv1 = bout[(nt0 + 1) * 16 + n];
#pragma unroll
  for (int r = 0; r < 8; ++r) {
    logits[(size_t)(mb + r) * VT_ + nt0 * 16 + n]       = acc0[r] + bv0;
    logits[(size_t)(mb + r) * VT_ + (nt0 + 1) * 16 + n] = acc1[r] + bv1;
  }
}

// ---------------- row log-softmax over 32000, write out[b, t, :] ----------------
__global__ __launch_bounds__(256)
void k_logsoftmax(const float* logits, float* out, int t) {
  __shared__ float red[256];
  int b = blockIdx.x, tid = threadIdx.x;
  const float* row = logits + (size_t)b * VT_;
  float mx = -INFINITY;
  for (int v = tid; v < VT_; v += 256) mx = fmaxf(mx, row[v]);
  red[tid] = mx; __syncthreads();
  for (int s = 128; s > 0; s >>= 1) {
    if (tid < s) red[tid] = fmaxf(red[tid], red[tid + s]);
    __syncthreads();
  }
  mx = red[0]; __syncthreads();
  float sum = 0.f;
  for (int v = tid; v < VT_; v += 256) sum += __expf(row[v] - mx);
  red[tid] = sum; __syncthreads();
  for (int s = 128; s > 0; s >>= 1) {
    if (tid < s) red[tid] += red[tid + s];
    __syncthreads();
  }
  float lse = mx + __logf(red[0]);
  float* orow = out + ((size_t)b * LT_ + t) * VT_;
  for (int v = tid; v < VT_; v += 256) orow[v] = row[v] - lse;
}

// ---------------- host ----------------

extern "C" void kernel_launch(void* const* d_in, const int* in_sizes, int n_in,
                              void* d_out, int out_size, void* d_ws, size_t ws_size,
                              hipStream_t stream) {
  const int*   src       = (const int*)d_in[0];
  const int*   tgt       = (const int*)d_in[1];
  const float* enc_embed = (const float*)d_in[2];
  const float* enc_Wx    = (const float*)d_in[3];
  const float* enc_Wh    = (const float*)d_in[4];
  const float* enc_b     = (const float*)d_in[5];
  const float* dec_embed = (const float*)d_in[6];
  const float* Wq        = (const float*)d_in[7];
  const float* Wk        = (const float*)d_in[8];
  const float* v_attn    = (const float*)d_in[9];
  const float* dec_Wx    = (const float*)d_in[10];
  const float* dec_Wh    = (const float*)d_in[11];
  const float* dec_b     = (const float*)d_in[12];
  const float* Wout      = (const float*)d_in[13];
  const float* bout      = (const float*)d_in[14];
  float* out = (float*)d_out;

  // carve workspace
  char* p = (char*)d_ws;
  auto take = [&](size_t bytes) {
    void* r = (void*)p;
    p += (bytes + 255) & ~(size_t)255;
    return r;
  };
  ush* bWq    = (ush*)take((size_t)H_ * H_ * 2);
  ush* bWk    = (ush*)take((size_t)H_ * H_ * 2);
  ush* beWx   = (ush*)take((size_t)E_ * G4_ * 2);
  ush* beWh   = (ush*)take((size_t)H_ * G4_ * 2);
  ush* bdWx   = (ush*)take((size_t)(E_ + H_) * G4_ * 2);
  ush* bdWh   = (ush*)take((size_t)H_ * G4_ * 2);
  ush* bWout  = (ush*)take((size_t)(2 * H_) * VT_ * 2);
  ush* apk    = (ush*)take((size_t)32 * 512 * 2);          // packed [h,ctx] A tiles
  float* h_buf  = (float*)take((size_t)2 * B_ * H_ * 4);   // ping-pong [2][B][H]
  float* c_buf  = (float*)take((size_t)B_ * H_ * 4);
  float* enc_hs = (float*)take((size_t)LS_ * B_ * H_ * 4); // [Ls][B][H]
  float* k_proj = (float*)take((size_t)LS_ * B_ * H_ * 4);
  float* qbuf   = (float*)take((size_t)B_ * H_ * 4);
  float* ctx    = (float*)take((size_t)B_ * H_ * 4);
  float* logits = (float*)take((size_t)B_ * VT_ * 4);

  auto pack = [&](const float* s, ush* d, int K, int N) {
    int n = K * N;
    k_pack<<<(n + 255) / 256, 256, 0, stream>>>(s, d, K, N);
  };
  pack(Wq, bWq, H_, H_);
  pack(Wk, bWk, H_, H_);
  pack(enc_Wx, beWx, E_, G4_);
  pack(enc_Wh, beWh, H_, G4_);
  pack(dec_Wx, bdWx, E_ + H_, G4_);
  pack(dec_Wh, bdWh, H_, G4_);
  pack(Wout, bWout, 2 * H_, VT_);

  k_zero<<<(2 * B_ * H_ + 255) / 256, 256, 0, stream>>>(h_buf, 2 * B_ * H_);
  k_zero<<<(B_ * H_ + 255) / 256, 256, 0, stream>>>(c_buf, B_ * H_);

  // encoder: 128 serial steps, h ping-pongs
  for (int t = 0; t < LS_; ++t) {
    k_enc_step<<<H_ / 16, 128, 0, stream>>>(
        src, enc_embed, beWx, beWh, enc_b,
        h_buf + (t & 1) * B_ * H_, h_buf + ((t + 1) & 1) * B_ * H_,
        c_buf, enc_hs, t);
  }

  // k_proj = enc_hs @ Wk  ([Ls*B, H] x [H, H])
  k_mm<<<dim3(LS_ * B_ / 16, H_ / 64), 128, 0, stream>>>(enc_hs, bWk, k_proj,
                                                         LS_ * B_, H_, H_);

  // decoder: 64 serial steps (h continues ping-pong from encoder: final h in buf 0)
  for (int t = 0; t < LT_; ++t) {
    const float* h_prev = h_buf + (t & 1) * B_ * H_;
    float* h_next = h_buf + ((t + 1) & 1) * B_ * H_;
    k_mm<<<dim3(1, H_ / 64), 128, 0, stream>>>(h_prev, bWq, qbuf, B_, H_, H_);
    k_attn<<<B_, 128, 0, stream>>>(src, k_proj, qbuf, v_attn, enc_hs, ctx);
    k_dec_step<<<H_ / 16, 128, 0, stream>>>(tgt, dec_embed, ctx, bdWx, bdWh, dec_b,
                                            h_prev, h_next, c_buf, t);
    k_packA_hx<<<64, 256, 0, stream>>>(h_next, ctx, apk);
    k_logits<<<VT_ / (16 * 8), 128, 0, stream>>>(apk, bWout, bout, logits);
    k_logsoftmax<<<B_, 256, 0, stream>>>(logits, out, t);
  }
}
